// GCN_28759101014034
// MI455X (gfx1250) — matrix operand order repaired and checked by reading, not compiled, not run on hardware
//
#include <hip/hip_runtime.h>
#include <hip/hip_bf16.h>

// ---------------------------------------------------------------------------
// Types for CDNA5 WMMA
// ---------------------------------------------------------------------------
typedef __attribute__((ext_vector_type(16))) __bf16 v16bf;
typedef __attribute__((ext_vector_type(8)))  float  v8f;

__device__ __forceinline__ int imin(int a, int b) { return a < b ? a : b; }

// ---------------------------------------------------------------------------
// fill: grid-stride constant fill
// ---------------------------------------------------------------------------
__global__ void k_fill(float* __restrict__ p, size_t n, float v) {
    size_t i = (size_t)blockIdx.x * blockDim.x + threadIdx.x;
    size_t stride = (size_t)gridDim.x * blockDim.x;
    for (; i < n; i += stride) p[i] = v;
}

// ---------------------------------------------------------------------------
// degree count: deg[dst[e]] += 1  (deg pre-filled with 1.0 for self-loops)
// ---------------------------------------------------------------------------
__global__ void k_count_deg(const long long* __restrict__ dst, float* __restrict__ deg, int E) {
    int e = blockIdx.x * blockDim.x + threadIdx.x;
    if (e < E) atomicAdd(&deg[dst[e]], 1.0f);
}

// dinv[i] = rsqrt(deg[i])  (deg >= 1 always)
__global__ void k_rsqrt(float* __restrict__ d, int n) {
    int i = blockIdx.x * blockDim.x + threadIdx.x;
    if (i < n) d[i] = rsqrtf(d[i]);
}

// ---------------------------------------------------------------------------
// One-shot: swizzle W[K x NC] (fp32, row-major) into bf16 B-fragment layout.
// Fragment layout (ISA 7.12.2, 16-bit B 32x16):
//   Wf[ ((kk*(NC/16) + c)*32 + lane)*16 + i ] = bf16( W[kk*32 + (i&7) + 16*(i>>3) + 8*(lane>>4)][c*16 + (lane&15)] )
// so in the GEMM each lane's B fragment is 32 contiguous bytes.
// ---------------------------------------------------------------------------
__global__ void k_swizzleW(const float* __restrict__ W, __bf16* __restrict__ Wf,
                           int K, int NC) {
    const int d = blockIdx.x * blockDim.x + threadIdx.x;
    if (d >= K * NC) return;
    const int i    = d & 15;
    const int l    = (d >> 4) & 31;
    const int rest = d >> 9;                 // kk*(NC/16) + c
    const int nt   = NC >> 4;
    const int c    = rest % nt;
    const int kk   = rest / nt;
    const int kg   = (kk << 5) + (i & 7) + ((i >> 3) << 4) + ((l >> 4) << 3);
    Wf[d] = (__bf16)W[(size_t)kg * NC + (c << 4) + (l & 15)];
}

// ---------------------------------------------------------------------------
// WMMA GEMM: Out[nRows x NC] = A[nRows x K] @ W[K x NC]
// bf16 multiply / f32 accumulate via v_wmma_f32_16x16x32_bf16.
//
// Block: WAVES waves (blockDim = 32*WAVES); each wave owns TPW = NC/(16*WAVES)
// output column tiles, grid.x = ceil(nRows/16) row tiles.
// Fully unrolled over K/32 steps; A fragment loaded once per k-step and reused
// across TPW wmmas. B fragments come pre-swizzled from global (L1/L2 hot).
//
// A-fragment (16x32 bf16, ISA 7.12.2): lane m=l&15, h=l>>4:
//   elements 0..7  = K = k0 + 8h .. k0+8h+7
//   elements 8..15 = K = k0 + 16 + 8h .. k0+16+8h+7
// D layout: VGPR r holds row m = r + 8*h, col n = lane&15.
// ---------------------------------------------------------------------------
template <int K, int NC, int WAVES>
__global__ void k_gemm_wmma(const float* __restrict__ A,
                            const __bf16* __restrict__ Wf,
                            float* __restrict__ Out, int nRows) {
    constexpr int TPW = NC / (16 * WAVES);   // col tiles per wave
    constexpr int NT  = NC / 16;             // col tiles total

    const int lane = threadIdx.x & 31;
    const int wave = threadIdx.x >> 5;
    const int m    = lane & 15;
    const int hh   = lane >> 4;

    const int rowTile = blockIdx.x;
    const int rowLd   = imin(rowTile * 16 + m, nRows - 1);   // clamp loads
    const float* arow = A + (size_t)rowLd * K;

    v8f acc[TPW];
    const v8f vzero = {};
#pragma unroll
    for (int t = 0; t < TPW; ++t) acc[t] = vzero;

#pragma unroll
    for (int kk = 0; kk < K / 32; ++kk) {
        // ---- A fragment: two 8-contiguous-K runs per lane ------------------
        const float* pa = arow + (kk << 5) + (hh << 3);
        float4 f0 = *(const float4*)(pa + 0);
        float4 f1 = *(const float4*)(pa + 4);
        float4 f2 = *(const float4*)(pa + 16);
        float4 f3 = *(const float4*)(pa + 20);

        v16bf a;
        a[0]  = (__bf16)f0.x; a[1]  = (__bf16)f0.y; a[2]  = (__bf16)f0.z; a[3]  = (__bf16)f0.w;
        a[4]  = (__bf16)f1.x; a[5]  = (__bf16)f1.y; a[6]  = (__bf16)f1.z; a[7]  = (__bf16)f1.w;
        a[8]  = (__bf16)f2.x; a[9]  = (__bf16)f2.y; a[10] = (__bf16)f2.z; a[11] = (__bf16)f2.w;
        a[12] = (__bf16)f3.x; a[13] = (__bf16)f3.y; a[14] = (__bf16)f3.z; a[15] = (__bf16)f3.w;

        // ---- TPW wmmas reusing the same A fragment -------------------------
#pragma unroll
        for (int t = 0; t < TPW; ++t) {
            const int c = wave * TPW + t;
            const v16bf b = *(const v16bf*)(Wf + (((size_t)(kk * NT + c) << 5) + lane) * 16);
            acc[t] = __builtin_amdgcn_wmma_f32_16x16x32_bf16(
                /*neg_a=*/false, a, /*neg_b=*/false, b,
                /*c_mod=*/(short)0, acc[t], /*reuse_a=*/false, /*reuse_b=*/false);
        }
    }

    // ---- D write-out -------------------------------------------------------
#pragma unroll
    for (int t = 0; t < TPW; ++t) {
        const int c = wave * TPW + t;
#pragma unroll
        for (int r = 0; r < 8; ++r) {
            const int row = rowTile * 16 + r + (hh << 3);
            if (row < nRows)
                Out[(size_t)row * NC + (c << 4) + m] = acc[t][r];
        }
    }
}

// ---------------------------------------------------------------------------
// edge aggregation: one wave per edge; acc[dst] += P[src] * dinv[src]*dinv[dst]
// C = 256 -> 8 channels/lane, C = 128 -> 4 channels/lane (float4 gathers).
// Feature buffers are L2-resident (51/26 MB vs 192 MB L2).
// ---------------------------------------------------------------------------
__global__ void k_agg_edges(const float* __restrict__ P,
                            const long long* __restrict__ src,
                            const long long* __restrict__ dst,
                            const float* __restrict__ dinv,
                            float* __restrict__ acc, int E, int C) {
    const int gw   = (blockIdx.x * blockDim.x + threadIdx.x) >> 5;
    const int lane = threadIdx.x & 31;
    if (gw >= E) return;

    const long long s = src[gw];
    const long long d = dst[gw];
    const float nrm = dinv[s] * dinv[d];

    const int per = C >> 5;                    // 8 or 4
    const float* prow = P   + (size_t)s * C + lane * per;
    float*       arow = acc + (size_t)d * C + lane * per;

    for (int j = 0; j < per; j += 4) {
        float4 v = *(const float4*)(prow + j);
        atomicAdd(arow + j + 0, v.x * nrm);
        atomicAdd(arow + j + 1, v.y * nrm);
        atomicAdd(arow + j + 2, v.z * nrm);
        atomicAdd(arow + j + 3, v.w * nrm);
    }
}

// ---------------------------------------------------------------------------
// finalize: acc = act( acc + P*dinv^2 (self-loop) + bias )
// act: 0 = relu, 1 = sigmoid
// ---------------------------------------------------------------------------
__global__ void k_finalize(const float* __restrict__ P,
                           const float* __restrict__ dinv,
                           const float* __restrict__ bias,
                           float* __restrict__ acc,
                           int N, int C, int act) {
    const size_t idx = (size_t)blockIdx.x * blockDim.x + threadIdx.x;
    if (idx >= (size_t)N * C) return;
    const int i = (int)(idx / C);
    const int c = (int)(idx % C);
    const float di = dinv[i];
    float v = acc[idx] + P[idx] * di * di + bias[c];
    acc[idx] = act ? (1.0f / (1.0f + expf(-v))) : fmaxf(v, 0.0f);
}

// ---------------------------------------------------------------------------
// launch
// ---------------------------------------------------------------------------
extern "C" void kernel_launch(void* const* d_in, const int* in_sizes, int n_in,
                              void* d_out, int out_size, void* d_ws, size_t ws_size,
                              hipStream_t stream) {
    const int IN_C = 512, HID = 256, CLS = 128;

    const float*     x    = (const float*)d_in[0];
    const float*     W1   = (const float*)d_in[1];
    const float*     b1   = (const float*)d_in[2];
    const float*     W2   = (const float*)d_in[3];
    const float*     b2   = (const float*)d_in[4];
    const long long* eidx = (const long long*)d_in[5];   // int64, shape (2,E)

    const int N = in_sizes[0] / IN_C;    // 50000
    const int E = in_sizes[5] / 2;       // 800000
    const long long* src = eidx;
    const long long* dst = eidx + E;

    float* w    = (float*)d_ws;
    float* dinv = w;                       // N floats
    float* P1   = dinv + N;                // N*256
    float* H    = P1 + (size_t)N * HID;    // N*256
    float* P2   = H  + (size_t)N * HID;    // N*128
    __bf16* W1f = (__bf16*)(P2 + (size_t)N * CLS);   // 512*256 bf16 (32B-aligned offset)
    __bf16* W2f = W1f + (size_t)IN_C * HID;          // 256*128 bf16
    float* out  = (float*)d_out;           // N*128

    // 0) one-shot weight swizzles into B-fragment layout (bf16)
    k_swizzleW<<<(IN_C * HID + 255) / 256, 256, 0, stream>>>(W1, W1f, IN_C, HID);
    k_swizzleW<<<(HID * CLS + 255) / 256, 256, 0, stream>>>(W2, W2f, HID, CLS);

    // 1) degrees -> dinv
    k_fill<<<512, 256, 0, stream>>>(dinv, (size_t)N, 1.0f);
    k_count_deg<<<(E + 255) / 256, 256, 0, stream>>>(dst, dinv, E);
    k_rsqrt<<<(N + 255) / 256, 256, 0, stream>>>(dinv, N);

    // 2) P1 = X @ W1 (WMMA bf16, 4 col tiles per wave, fully unrolled K)
    k_gemm_wmma<512, 256, 4><<<(N + 15) / 16, 128, 0, stream>>>(x, W1f, P1, N);

    // 3) H = scatter-add of P1 over edges, then relu(H + self-loop + b1)
    k_fill<<<2048, 256, 0, stream>>>(H, (size_t)N * HID, 0.0f);
    k_agg_edges<<<(E + 7) / 8, 256, 0, stream>>>(P1, src, dst, dinv, H, E, HID);
    k_finalize<<<(int)(((size_t)N * HID + 255) / 256), 256, 0, stream>>>(P1, dinv, b1, H, N, HID, 0);

    // 4) P2 = H @ W2 (WMMA bf16, 2 col tiles per wave)
    k_gemm_wmma<256, 128, 4><<<(N + 15) / 16, 128, 0, stream>>>(H, W2f, P2, N);

    // 5) out = sigmoid( scatter-add of P2 + self-loop + b2 )
    k_fill<<<2048, 256, 0, stream>>>(out, (size_t)N * CLS, 0.0f);
    k_agg_edges<<<(E + 7) / 8, 256, 0, stream>>>(P2, src, dst, dinv, out, E, CLS);
    k_finalize<<<(int)(((size_t)N * CLS + 255) / 256), 256, 0, stream>>>(P2, dinv, b2, out, N, CLS, 1);
}
